// MapAttention_dynamic_pos_43095701848464
// MI455X (gfx1250) — matrix operand order repaired and checked by reading, not compiled, not run on hardware
//
#include <hip/hip_runtime.h>
#include <hip/hip_bf16.h>

// Problem sizes (fixed by the reference)
#define B_SZ 8
#define NQ   2048
#define NKV  2048
#define DQ   512
#define DIN  512

typedef __attribute__((ext_vector_type(16))) _Float16 v16h;
typedef __attribute__((ext_vector_type(8)))  _Float16 v8h;
typedef __attribute__((ext_vector_type(8)))  float    v8f;
typedef __attribute__((ext_vector_type(4)))  float    v4f;

// ---------------------------------------------------------------------------
// WMMA fragment helpers (CDNA5 wave32 layouts, CDNA5 ISA §7.12.2)
//
// A (16x32 f16, MxK): lane L holds row M=L&15.
//   lanes 0-15 : elems 0..7 = K 0..7,   elems 8..15 = K 16..23
//   lanes 16-31: elems 0..7 = K 8..15,  elems 8..15 = K 24..31
// B (32x16 f16, KxN): lane L holds col N=L&15.
//   lanes 0-15 : elems 0..7 = K 0..7,   elems 8..15 = K 16..23
//   lanes 16-31: elems 0..7 = K 8..15,  elems 8..15 = K 24..31
// C/D (16x16 f32): lane L holds col N=L&15, rows M = r + (L>=16 ? 8 : 0).
// ---------------------------------------------------------------------------

__device__ __forceinline__ v16h load_a_f16(const _Float16* __restrict__ A,
                                           int lda, int lane) {
  int m  = lane & 15;
  int kb = (lane & 16) ? 8 : 0;
  const _Float16* p = A + (size_t)m * lda + kb;
  v8h lo = *(const v8h*)(p);       // K = kb .. kb+7
  v8h hi = *(const v8h*)(p + 16);  // K = kb+16 .. kb+23
  v16h r;
#pragma unroll
  for (int i = 0; i < 8; ++i) { r[i] = lo[i]; r[i + 8] = hi[i]; }
  return r;
}

// Same A fragment but source is fp32 (convert in-register, used for attn).
__device__ __forceinline__ v16h load_a_f32(const float* __restrict__ A,
                                           int lda, int lane) {
  int m  = lane & 15;
  int kb = (lane & 16) ? 8 : 0;
  const float* p = A + (size_t)m * lda + kb;
  v4f a0 = *(const v4f*)(p);
  v4f a1 = *(const v4f*)(p + 4);
  v4f b0 = *(const v4f*)(p + 16);
  v4f b1 = *(const v4f*)(p + 20);
  v16h r;
#pragma unroll
  for (int i = 0; i < 4; ++i) {
    r[i]      = (_Float16)a0[i];
    r[i + 4]  = (_Float16)a1[i];
    r[i + 8]  = (_Float16)b0[i];
    r[i + 12] = (_Float16)b1[i];
  }
  return r;
}

// B fragment when B[k][n] == Bt[n][k] with Bt row-major (pitch ldb):
// per-lane load is K kb..kb+7 and kb+16..kb+23 (kb = 0 or 8) -> two b128 loads.
__device__ __forceinline__ v16h load_bt_f16(const _Float16* __restrict__ Bt,
                                            int ldb, int lane) {
  int n  = lane & 15;
  int kb = (lane & 16) ? 8 : 0;
  const _Float16* p = Bt + (size_t)n * ldb + kb;
  v8h lo = *(const v8h*)(p);
  v8h hi = *(const v8h*)(p + 16);
  v16h r;
#pragma unroll
  for (int i = 0; i < 8; ++i) { r[i] = lo[i]; r[i + 8] = hi[i]; }
  return r;
}

#define WMMA_F16(A, Bv, C) \
  __builtin_amdgcn_wmma_f32_16x16x32_f16(false, (A), false, (Bv), (short)0, (C), false, false)

// ---------------------------------------------------------------------------
// Elementwise fp32 -> fp16 conversion (grid-stride)
// ---------------------------------------------------------------------------
__global__ __launch_bounds__(256) void cvt_f32_to_f16(const float* __restrict__ src,
                                                      _Float16* __restrict__ dst,
                                                      size_t n) {
  size_t i      = (size_t)blockIdx.x * blockDim.x + threadIdx.x;
  size_t stride = (size_t)gridDim.x * blockDim.x;
  for (; i < n; i += stride) dst[i] = (_Float16)src[i];
}

// ---------------------------------------------------------------------------
// x[b, t, i] (f32) -> xT[b, i, t] (f16), 32x32 LDS tiles, 256 threads (32x8)
// ---------------------------------------------------------------------------
__global__ __launch_bounds__(256) void transpose_cvt(const float* __restrict__ x,
                                                     _Float16* __restrict__ xT) {
  __shared__ _Float16 tile[32][33];
  int b  = blockIdx.z;
  int t0 = blockIdx.x * 32;
  int i0 = blockIdx.y * 32;
  const float* xb  = x  + (size_t)b * NKV * DIN;
  _Float16*    xTb = xT + (size_t)b * DIN * NKV;
  int tx = threadIdx.x & 31;
  int ty = threadIdx.x >> 5;
#pragma unroll
  for (int r = 0; r < 32; r += 8)
    tile[ty + r][tx] = (_Float16)xb[(size_t)(t0 + ty + r) * DIN + i0 + tx];
  __syncthreads();
#pragma unroll
  for (int r = 0; r < 32; r += 8)
    xTb[(size_t)(i0 + ty + r) * NKV + t0 + tx] = tile[tx][ty + r];
}

// ---------------------------------------------------------------------------
// GEMM 1: qp[b,q,e] = sum_d q16[b,q,d] * Wp16[e,d] + b_proj[e]   (f16 out)
// Block tile 128Mx128N, wave tile 32Mx64N.
// ---------------------------------------------------------------------------
__global__ __launch_bounds__(256) void gemm_proj_q(const _Float16* __restrict__ q16,
                                                   const _Float16* __restrict__ Wp16,
                                                   const float* __restrict__ bias,
                                                   _Float16* __restrict__ qp16) {
  int b    = blockIdx.z;
  int lane = threadIdx.x & 31;
  int wid  = threadIdx.x >> 5;
  int m0 = blockIdx.x * 128 + (wid & 3) * 32;
  int n0 = blockIdx.y * 128 + (wid >> 2) * 64;
  const _Float16* A = q16 + (size_t)b * NQ * DQ;

  v8f acc[2][4] = {};
  for (int k0 = 0; k0 < DQ; k0 += 32) {
    v16h a0 = load_a_f16(A + (size_t)m0 * DQ + k0, DQ, lane);
    v16h a1 = load_a_f16(A + (size_t)(m0 + 16) * DQ + k0, DQ, lane);
#pragma unroll
    for (int j = 0; j < 4; ++j) {
      v16h bf = load_bt_f16(Wp16 + (size_t)(n0 + j * 16) * DQ + k0, DQ, lane);
      acc[0][j] = WMMA_F16(a0, bf, acc[0][j]);
      acc[1][j] = WMMA_F16(a1, bf, acc[1][j]);
    }
  }

  _Float16* C = qp16 + (size_t)b * NQ * DQ;
  int n  = lane & 15;
  int mb = (lane & 16) ? 8 : 0;
#pragma unroll
  for (int i = 0; i < 2; ++i)
#pragma unroll
    for (int j = 0; j < 4; ++j) {
      float bv = bias[n0 + j * 16 + n];
#pragma unroll
      for (int r = 0; r < 8; ++r)
        C[(size_t)(m0 + i * 16 + mb + r) * DQ + n0 + j * 16 + n] =
            (_Float16)(acc[i][j][r] + bv);
    }
}

// ---------------------------------------------------------------------------
// GEMM 2: kk[b,t,e] = sum_i x16[b,t,i] * Wx16[e,i] + b_px[e] + key[t,e]
// ---------------------------------------------------------------------------
__global__ __launch_bounds__(256) void gemm_proj_k(const _Float16* __restrict__ x16,
                                                   const _Float16* __restrict__ Wx16,
                                                   const float* __restrict__ bias,
                                                   const float* __restrict__ key,
                                                   _Float16* __restrict__ kk16) {
  int b    = blockIdx.z;
  int lane = threadIdx.x & 31;
  int wid  = threadIdx.x >> 5;
  int m0 = blockIdx.x * 128 + (wid & 3) * 32;
  int n0 = blockIdx.y * 128 + (wid >> 2) * 64;
  const _Float16* A = x16 + (size_t)b * NKV * DIN;

  v8f acc[2][4] = {};
  for (int k0 = 0; k0 < DIN; k0 += 32) {
    v16h a0 = load_a_f16(A + (size_t)m0 * DIN + k0, DIN, lane);
    v16h a1 = load_a_f16(A + (size_t)(m0 + 16) * DIN + k0, DIN, lane);
#pragma unroll
    for (int j = 0; j < 4; ++j) {
      v16h bf = load_bt_f16(Wx16 + (size_t)(n0 + j * 16) * DIN + k0, DIN, lane);
      acc[0][j] = WMMA_F16(a0, bf, acc[0][j]);
      acc[1][j] = WMMA_F16(a1, bf, acc[1][j]);
    }
  }

  _Float16* C = kk16 + (size_t)b * NKV * DQ;
  int n  = lane & 15;
  int mb = (lane & 16) ? 8 : 0;
#pragma unroll
  for (int i = 0; i < 2; ++i)
#pragma unroll
    for (int j = 0; j < 4; ++j) {
      int gn   = n0 + j * 16 + n;
      float bv = bias[gn];
#pragma unroll
      for (int r = 0; r < 8; ++r) {
        int gm = m0 + i * 16 + mb + r;
        C[(size_t)gm * DQ + gn] =
            (_Float16)(acc[i][j][r] + bv + key[(size_t)gm * DQ + gn]);
      }
    }
}

// ---------------------------------------------------------------------------
// GEMM 3: score[b,q,t] = sum_e qp16[b,q,e] * kk16[b,t,e]   (f32 out -> attn)
// ---------------------------------------------------------------------------
__global__ __launch_bounds__(256) void gemm_score(const _Float16* __restrict__ qp16,
                                                  const _Float16* __restrict__ kk16,
                                                  float* __restrict__ attn) {
  int b    = blockIdx.z;
  int lane = threadIdx.x & 31;
  int wid  = threadIdx.x >> 5;
  int m0 = blockIdx.x * 128 + (wid & 3) * 32;
  int n0 = blockIdx.y * 128 + (wid >> 2) * 64;
  const _Float16* A  = qp16 + (size_t)b * NQ * DQ;
  const _Float16* Bt = kk16 + (size_t)b * NKV * DQ;

  v8f acc[2][4] = {};
  for (int k0 = 0; k0 < DQ; k0 += 32) {
    v16h a0 = load_a_f16(A + (size_t)m0 * DQ + k0, DQ, lane);
    v16h a1 = load_a_f16(A + (size_t)(m0 + 16) * DQ + k0, DQ, lane);
#pragma unroll
    for (int j = 0; j < 4; ++j) {
      v16h bf = load_bt_f16(Bt + (size_t)(n0 + j * 16) * DQ + k0, DQ, lane);
      acc[0][j] = WMMA_F16(a0, bf, acc[0][j]);
      acc[1][j] = WMMA_F16(a1, bf, acc[1][j]);
    }
  }

  float* C = attn + (size_t)b * NQ * NKV;
  int n  = lane & 15;
  int mb = (lane & 16) ? 8 : 0;
#pragma unroll
  for (int i = 0; i < 2; ++i)
#pragma unroll
    for (int j = 0; j < 4; ++j)
#pragma unroll
      for (int r = 0; r < 8; ++r)
        C[(size_t)(m0 + i * 16 + mb + r) * NKV + n0 + j * 16 + n] = acc[i][j][r];
}

// ---------------------------------------------------------------------------
// Row softmax over NKV=2048, in place. One 256-thread block per (b,q) row.
// ---------------------------------------------------------------------------
__global__ __launch_bounds__(256) void softmax_rows(float* __restrict__ attn) {
  __shared__ float red[8];
  float* p = attn + (size_t)blockIdx.x * NKV;
  int tid  = threadIdx.x;
  int lane = tid & 31;
  int wid  = tid >> 5;

  float v[8];
  float m = -__builtin_inff();
#pragma unroll
  for (int j = 0; j < 8; ++j) {
    v[j] = p[tid + j * 256];
    m = fmaxf(m, v[j]);
  }
#pragma unroll
  for (int o = 16; o >= 1; o >>= 1) m = fmaxf(m, __shfl_xor(m, o, 32));
  if (lane == 0) red[wid] = m;
  __syncthreads();
#pragma unroll
  for (int w = 0; w < 8; ++w) m = fmaxf(m, red[w]);

  float s = 0.f;
#pragma unroll
  for (int j = 0; j < 8; ++j) {
    v[j] = __expf(v[j] - m);
    s += v[j];
  }
#pragma unroll
  for (int o = 16; o >= 1; o >>= 1) s += __shfl_xor(s, o, 32);
  __syncthreads();
  if (lane == 0) red[wid] = s;
  __syncthreads();
  s = 0.f;
#pragma unroll
  for (int w = 0; w < 8; ++w) s += red[w];

  float inv = 1.0f / s;
#pragma unroll
  for (int j = 0; j < 8; ++j) p[tid + j * 256] = v[j] * inv;
}

// ---------------------------------------------------------------------------
// GEMM 4: out[b,q,i] = sum_t attn[b,q,t] * xT16[b,i,t]   (A is fp32, cvt'd)
// ---------------------------------------------------------------------------
__global__ __launch_bounds__(256) void gemm_out(const float* __restrict__ attn,
                                                const _Float16* __restrict__ xT16,
                                                float* __restrict__ out) {
  int b    = blockIdx.z;
  int lane = threadIdx.x & 31;
  int wid  = threadIdx.x >> 5;
  int m0 = blockIdx.x * 128 + (wid & 3) * 32;
  int n0 = blockIdx.y * 128 + (wid >> 2) * 64;
  const float*    A  = attn + (size_t)b * NQ * NKV;
  const _Float16* Bt = xT16 + (size_t)b * DIN * NKV;

  v8f acc[2][4] = {};
  for (int k0 = 0; k0 < NKV; k0 += 32) {
    v16h a0 = load_a_f32(A + (size_t)m0 * NKV + k0, NKV, lane);
    v16h a1 = load_a_f32(A + (size_t)(m0 + 16) * NKV + k0, NKV, lane);
#pragma unroll
    for (int j = 0; j < 4; ++j) {
      v16h bf = load_bt_f16(Bt + (size_t)(n0 + j * 16) * NKV + k0, NKV, lane);
      acc[0][j] = WMMA_F16(a0, bf, acc[0][j]);
      acc[1][j] = WMMA_F16(a1, bf, acc[1][j]);
    }
  }

  float* C = out + (size_t)b * NQ * DIN;
  int n  = lane & 15;
  int mb = (lane & 16) ? 8 : 0;
#pragma unroll
  for (int i = 0; i < 2; ++i)
#pragma unroll
    for (int j = 0; j < 4; ++j)
#pragma unroll
      for (int r = 0; r < 8; ++r)
        C[(size_t)(m0 + i * 16 + mb + r) * DIN + n0 + j * 16 + n] = acc[i][j][r];
}

// ---------------------------------------------------------------------------
// Host launcher
// ---------------------------------------------------------------------------
extern "C" void kernel_launch(void* const* d_in, const int* in_sizes, int n_in,
                              void* d_out, int out_size, void* d_ws, size_t ws_size,
                              hipStream_t stream) {
  const float* query  = (const float*)d_in[0];  // [B, NQ, DQ]
  const float* key    = (const float*)d_in[1];  // [NKV, DQ]
  const float* x      = (const float*)d_in[2];  // [B, NKV, DIN]
  const float* W_proj = (const float*)d_in[3];  // [DQ, DQ]
  const float* b_proj = (const float*)d_in[4];  // [DQ]
  const float* W_px   = (const float*)d_in[5];  // [DQ, DIN]
  const float* b_px   = (const float*)d_in[6];  // [DQ]

  float* out  = (float*)d_out;                         // [B, NQ, DIN]
  float* attn = out + (size_t)B_SZ * NQ * DIN;         // [B, NQ, NKV]

  const size_t NX = (size_t)B_SZ * NQ * DQ;            // 8,388,608 elements
  _Float16* w    = (_Float16*)d_ws;
  _Float16* q16  = w;                 // [B, NQ, DQ]
  _Float16* x16  = q16 + NX;          // [B, NKV, DIN]
  _Float16* xT16 = x16 + NX;          // [B, DIN, NKV]
  _Float16* qp16 = xT16 + NX;         // [B, NQ, DQ]
  _Float16* kk16 = qp16 + NX;         // [B, NKV, DQ]
  _Float16* Wp16 = kk16 + NX;         // [DQ, DQ]
  _Float16* Wx16 = Wp16 + (size_t)DQ * DQ;  // [DQ, DIN]

  // 1) fp16 staging
  cvt_f32_to_f16<<<8192, 256, 0, stream>>>(query, q16, NX);
  cvt_f32_to_f16<<<8192, 256, 0, stream>>>(x, x16, NX);
  cvt_f32_to_f16<<<1024, 256, 0, stream>>>(W_proj, Wp16, (size_t)DQ * DQ);
  cvt_f32_to_f16<<<1024, 256, 0, stream>>>(W_px, Wx16, (size_t)DQ * DIN);
  transpose_cvt<<<dim3(NKV / 32, DIN / 32, B_SZ), 256, 0, stream>>>(x, xT16);

  // 2) projections
  gemm_proj_q<<<dim3(NQ / 128, DQ / 128, B_SZ), 256, 0, stream>>>(q16, Wp16, b_proj, qp16);
  gemm_proj_k<<<dim3(NKV / 128, DQ / 128, B_SZ), 256, 0, stream>>>(x16, Wx16, b_px, key, kk16);

  // 3) scores -> attn region of d_out
  gemm_score<<<dim3(NQ / 128, NKV / 128, B_SZ), 256, 0, stream>>>(qp16, kk16, attn);

  // 4) softmax in place
  softmax_rows<<<B_SZ * NQ, 256, 0, stream>>>(attn);

  // 5) out = attn @ x
  gemm_out<<<dim3(NQ / 128, DIN / 128, B_SZ), 256, 0, stream>>>(attn, xT16, out);
}
